// EncoderBlock_4922032521403
// MI455X (gfx1250) — matrix operand order repaired and checked by reading, not compiled, not run on hardware
//
#include <hip/hip_runtime.h>
#include <hip/hip_bf16.h>

#define BB  16
#define LL  512
#define CC  512
#define HH  4
#define HSZ 128
#define HIDD 2048

typedef __attribute__((ext_vector_type(16))) __bf16 v16bf;
typedef __attribute__((ext_vector_type(8)))  float  v8f;

union Frag16 { uint4 q[2]; v16bf v; };

__device__ __forceinline__ unsigned short f2bf(float f) {
  unsigned int u = __float_as_uint(f);
  u += 0x7FFFu + ((u >> 16) & 1u);          // round-to-nearest-even
  return (unsigned short)(u >> 16);
}

// ---------------------------------------------------------------------------
// Generic strided-batched bf16 WMMA GEMM.
//   C[M,N] = alpha * A[M,K](row-major bf16) @ B[K,N]  (+ Cin if ACC)
//   B supplied COLUMN-major: Bt[n*K + k] == B[k][n]  -> contiguous frag loads.
//   Block: 256 thr = 8 waves arranged 4(M) x 2(N); block tile 128x128.
//   Wave tile 32x64: 2 A-frags x 4 B-frags = 8 f32 accumulators
//   -> 12 b128 loads per 8 WMMAs (1.5 loads/WMMA).
// ---------------------------------------------------------------------------
template<bool OUT_BF16, bool ACC>
__global__ __launch_bounds__(256)
void k_gemm(const unsigned short* __restrict__ A,
            const unsigned short* __restrict__ Bt,
            const float* __restrict__ Cin,
            void* __restrict__ Out,
            int M, int N, int K, float alpha,
            long strideA, long strideB,
            long strideC_outer, long strideC_inner, int nInner,
            int ldC)
{
  const int lane = threadIdx.x & 31;
  const int wave = threadIdx.x >> 5;
  const int wm = wave >> 1;                 // 0..3
  const int wn = wave & 1;                  // 0..1
  const int z  = blockIdx.z;

  const long offC = (long)(z / nInner) * strideC_outer
                  + (long)(z % nInner) * strideC_inner;
  const unsigned short* Ab = A  + (long)z * strideA;
  const unsigned short* Bb = Bt + (long)z * strideB;

  const int rowBase = blockIdx.y * 128 + wm * 32;
  const int colBase = blockIdx.x * 128 + wn * 64;

  // A 16x32 bf16 frag: row = lane&15; K run of 16 starting at (lane>=16)?16:0
  const int arow = rowBase + (lane & 15);
  const int akb  = (lane >> 4) << 4;
  // B 32x16 bf16 frag: col = lane&15; K run of 16 starting at (lane>=16)?16:0
  const int bcol = colBase + (lane & 15);
  const int bkb  = (lane >> 4) << 4;

  v8f acc[2][4];
  #pragma unroll
  for (int mi = 0; mi < 2; ++mi)
    #pragma unroll
    for (int ni = 0; ni < 4; ++ni) acc[mi][ni] = (v8f){};

  for (int k0 = 0; k0 < K; k0 += 32) {
    Frag16 af[2], bf[4];
    #pragma unroll
    for (int mi = 0; mi < 2; ++mi) {
      const uint4* p = (const uint4*)(Ab + (size_t)(arow + 16 * mi) * K + k0 + akb);
      af[mi].q[0] = p[0];
      af[mi].q[1] = p[1];                   // second K-half at +8 elems (+16B)
    }
    #pragma unroll
    for (int ni = 0; ni < 4; ++ni) {
      const uint4* p = (const uint4*)(Bb + (size_t)(bcol + 16 * ni) * K + k0 + bkb);
      bf[ni].q[0] = p[0];
      bf[ni].q[1] = p[1];
    }
    #pragma unroll
    for (int mi = 0; mi < 2; ++mi)
      #pragma unroll
      for (int ni = 0; ni < 4; ++ni)
        acc[mi][ni] = __builtin_amdgcn_wmma_f32_16x16x32_bf16(
            false, af[mi].v, false, bf[ni].v, (short)0, acc[mi][ni], false, false);
  }

  // C/D 16x16 f32: col = lane&15, VGPR r -> row r (lanes 0-15) / 8+r (lanes 16-31)
  const int crow = (lane >> 4) << 3;
  const int ccol = lane & 15;
  float*          of = (float*)Out;
  unsigned short* ob = (unsigned short*)Out;
  #pragma unroll
  for (int mi = 0; mi < 2; ++mi)
    #pragma unroll
    for (int ni = 0; ni < 4; ++ni) {
      #pragma unroll
      for (int r = 0; r < 8; ++r) {
        int  grow = rowBase + mi * 16 + crow + r;
        int  gcol = colBase + ni * 16 + ccol;
        long idx  = offC + (long)grow * ldC + gcol;
        float v = alpha * acc[mi][ni][r];
        if (ACC) v += Cin[idx];
        if (OUT_BF16) ob[idx] = f2bf(v); else of[idx] = v;
      }
    }
}

// --------------------------- elementwise kernels ---------------------------

__global__ __launch_bounds__(256)
void k_ln(const float* __restrict__ x, const float* __restrict__ g,
          const float* __restrict__ b, unsigned short* __restrict__ out)
{
  __shared__ float red[256];
  const int row = blockIdx.x;
  const float* xr = x + (size_t)row * CC;
  float s = 0.f;
  for (int i = threadIdx.x; i < CC; i += 256) s += xr[i];
  red[threadIdx.x] = s; __syncthreads();
  for (int o = 128; o > 0; o >>= 1) { if (threadIdx.x < o) red[threadIdx.x] += red[threadIdx.x + o]; __syncthreads(); }
  float mu = red[0] / CC; __syncthreads();
  float v = 0.f;
  for (int i = threadIdx.x; i < CC; i += 256) { float d = xr[i] - mu; v += d * d; }
  red[threadIdx.x] = v; __syncthreads();
  for (int o = 128; o > 0; o >>= 1) { if (threadIdx.x < o) red[threadIdx.x] += red[threadIdx.x + o]; __syncthreads(); }
  float rstd = rsqrtf(red[0] / CC + 1e-5f);
  unsigned short* orow = out + (size_t)row * CC;
  for (int i = threadIdx.x; i < CC; i += 256)
    orow[i] = f2bf((xr[i] - mu) * rstd * g[i] + b[i]);
}

__global__ void k_tr_f32_bf16(const float* __restrict__ src, unsigned short* __restrict__ dst,
                              int R, int Cc, long sStride, long dStride)
{
  long i = (long)blockIdx.x * blockDim.x + threadIdx.x;
  long total = (long)R * Cc;
  if (i >= total) return;
  int b = blockIdx.y;
  int r = (int)(i / Cc), c = (int)(i % Cc);
  dst[(long)b * dStride + (long)c * R + r] = f2bf(src[(long)b * sStride + i]);
}

__global__ void k_split_qkv(const float* __restrict__ qkv,
                            unsigned short* __restrict__ q,
                            unsigned short* __restrict__ k,
                            unsigned short* __restrict__ vT)
{
  long i = (long)blockIdx.x * blockDim.x + threadIdx.x;
  long total = (long)BB * LL * CC;
  if (i >= total) return;
  int b = (int)(i / ((long)LL * CC));
  int rem = (int)(i % ((long)LL * CC));
  int l = rem / CC, c = rem % CC;
  int h = c / HSZ, d = c % HSZ;
  const float* base = qkv + ((long)b * LL + l) * (3 * CC);
  long qi = (((long)b * HH + h) * LL + l) * HSZ + d;
  q[qi] = f2bf(base[c]);
  k[qi] = f2bf(base[CC + c]);
  vT[(((long)b * HH + h) * HSZ + d) * LL + l] = f2bf(base[2 * CC + c]);
}

__global__ void k_build_a(const float* __restrict__ rel,
                          unsigned short* __restrict__ a, unsigned short* __restrict__ aT)
{
  long i = (long)blockIdx.x * blockDim.x + threadIdx.x;
  long total = (long)BB * LL * LL;
  if (i >= total) return;
  int b = (int)(i / ((long)LL * LL));
  int rem = (int)(i % ((long)LL * LL));
  int r = rem / LL, c = rem % LL;
  float v = rel[i];
  float m = (v > 1.f && v < 9.f) ? 0.f : v;
  unsigned short o = f2bf(m != 0.f ? 1.f : 0.f);
  a[i] = o;
  aT[(long)b * LL * LL + (long)c * LL + r] = o;
}

__global__ void k_cvt_pair(const float* __restrict__ src,
                           unsigned short* __restrict__ d, unsigned short* __restrict__ dT)
{
  long i = (long)blockIdx.x * blockDim.x + threadIdx.x;
  long total = (long)BB * LL * LL;
  if (i >= total) return;
  int b = (int)(i / ((long)LL * LL));
  int rem = (int)(i % ((long)LL * LL));
  int r = rem / LL, c = rem % LL;
  unsigned short o = f2bf(src[i]);
  d[i] = o;
  dT[(long)b * LL * LL + (long)c * LL + r] = o;
}

__global__ void k_build_pe(const float* __restrict__ rel,
                           const float* __restrict__ p1, const float* __restrict__ p2,
                           unsigned char* __restrict__ pe)
{
  long i = (long)blockIdx.x * blockDim.x + threadIdx.x;
  long total = (long)BB * LL * LL;
  if (i >= total) return;
  int b = (int)(i / ((long)LL * LL));
  int rem = (int)(i % ((long)LL * LL));
  int r = rem / LL, c = rem % LL;
  float vr = rel[i];
  float vt = rel[(long)b * LL * LL + (long)c * LL + r];
  float av  = ((vr > 1.f && vr < 9.f) ? 0.f : vr) != 0.f ? 1.f : 0.f;
  float avT = ((vt > 1.f && vt < 9.f) ? 0.f : vt) != 0.f ? 1.f : 0.f;
  float eye = (r == c) ? 1.f : 0.f;
  long base = ((long)b * 4) * LL * LL + (long)r * LL + c;
  long chs  = (long)LL * LL;
  pe[base + 0 * chs] = (av    + eye) != 0.f;
  pe[base + 1 * chs] = (avT   + eye) != 0.f;
  pe[base + 2 * chs] = (p1[i] + eye) != 0.f;
  pe[base + 3 * chs] = (p2[i] + eye) != 0.f;
}

__global__ __launch_bounds__(256)
void k_softmax(const float* __restrict__ score, const unsigned char* __restrict__ pe,
               unsigned short* __restrict__ attn)
{
  __shared__ float red[256];
  int row = blockIdx.x;                         // over B*H*L
  int l = row % LL;
  int bh = row / LL;                            // b*H + h == b*4 + h (H==4)
  const float* sr = score + (size_t)row * LL;
  const unsigned char* pr = pe + ((size_t)bh * LL + l) * LL;
  float mx = -1e30f;
  for (int i = threadIdx.x; i < LL; i += 256)
    if (pr[i]) mx = fmaxf(mx, sr[i]);
  red[threadIdx.x] = mx; __syncthreads();
  for (int o = 128; o > 0; o >>= 1) { if (threadIdx.x < o) red[threadIdx.x] = fmaxf(red[threadIdx.x], red[threadIdx.x + o]); __syncthreads(); }
  mx = red[0]; __syncthreads();
  float s = 0.f;
  for (int i = threadIdx.x; i < LL; i += 256)
    if (pr[i]) s += expf(sr[i] - mx);
  red[threadIdx.x] = s; __syncthreads();
  for (int o = 128; o > 0; o >>= 1) { if (threadIdx.x < o) red[threadIdx.x] += red[threadIdx.x + o]; __syncthreads(); }
  float inv = 1.f / red[0];
  unsigned short* ar = attn + (size_t)row * LL;
  for (int i = threadIdx.x; i < LL; i += 256)
    ar[i] = pr[i] ? f2bf(expf(sr[i] - mx) * inv) : (unsigned short)0;
}

__global__ void k_relu_cvt(const float* __restrict__ h, unsigned short* __restrict__ hb, long total)
{
  long i = (long)blockIdx.x * blockDim.x + threadIdx.x;
  if (i >= total) return;
  float v = h[i];
  hb[i] = f2bf(v > 0.f ? v : 0.f);
}

// ---------------------------------------------------------------------------

extern "C" void kernel_launch(void* const* d_in, const int* in_sizes, int n_in,
                              void* d_out, int out_size, void* d_ws, size_t ws_size,
                              hipStream_t stream) {
  const float* x      = (const float*)d_in[0];
  const float* rel    = (const float*)d_in[1];
  const float* adj    = (const float*)d_in[2];
  const float* w_qkv  = (const float*)d_in[3];
  const float* w_proj = (const float*)d_in[4];
  const float* ln1g   = (const float*)d_in[5];
  const float* ln1b   = (const float*)d_in[6];
  const float* ln2g   = (const float*)d_in[7];
  const float* ln2b   = (const float*)d_in[8];
  const float* w_fc1  = (const float*)d_in[9];
  const float* w_gcn  = (const float*)d_in[10];
  const float* w_fc2  = (const float*)d_in[11];
  float* out = (float*)d_out;

  char* ws = (char*)d_ws;
  size_t off = 0;
  auto alloc = [&](size_t bytes) -> char* {
    char* p = ws + off;
    off = (off + bytes + 255) & ~(size_t)255;
    return p;
  };

  const long BLL = (long)BB * LL * LL;     // 4194304
  const long BLC = (long)BB * LL * CC;     // 4194304
  const long ROWS = (long)BB * LL;         // 8192

  unsigned short* xn     = (unsigned short*)alloc(BLC * 2);
  unsigned short* wqkvT  = (unsigned short*)alloc((size_t)CC * 3 * CC * 2);
  unsigned short* wprojT = (unsigned short*)alloc((size_t)CC * CC * 2);
  unsigned short* wfc1T  = (unsigned short*)alloc((size_t)CC * HIDD * 2);
  unsigned short* wgcnT  = (unsigned short*)alloc((size_t)CC * HIDD * 2);
  unsigned short* wfc2T  = (unsigned short*)alloc((size_t)CC * HIDD * 2);
  float*          qkvbuf = (float*)alloc((size_t)ROWS * 3 * CC * 4);
  unsigned short* qb     = (unsigned short*)alloc(BLC * 2);
  unsigned short* kb     = (unsigned short*)alloc(BLC * 2);
  unsigned short* vTb    = (unsigned short*)alloc(BLC * 2);
  unsigned short* ab     = (unsigned short*)alloc(BLL * 2);
  unsigned short* aTb    = (unsigned short*)alloc(BLL * 2);
  unsigned short* adjb   = (unsigned short*)alloc(BLL * 2);
  unsigned short* adjTb  = (unsigned short*)alloc(BLL * 2);
  float*          prod1  = (float*)alloc(BLL * 4);
  float*          prod2  = (float*)alloc(BLL * 4);
  unsigned char*  pe     = (unsigned char*)alloc((size_t)BB * 4 * LL * LL);
  float*          score  = (float*)alloc((size_t)BB * HH * LL * LL * 4);
  unsigned short* attnb  = (unsigned short*)alloc((size_t)BB * HH * LL * LL * 2);
  unsigned short* aout   = (unsigned short*)alloc(BLC * 2);
  float*          x1     = (float*)alloc(BLC * 4);
  unsigned short* xn2    = (unsigned short*)alloc(BLC * 2);
  float*          hbuf   = (float*)alloc((size_t)ROWS * HIDD * 4);
  float*          gbuf   = (float*)alloc((size_t)ROWS * HIDD * 4);
  unsigned short* gT     = (unsigned short*)alloc((size_t)BB * HIDD * LL * 2);
  unsigned short* hb     = (unsigned short*)alloc((size_t)ROWS * HIDD * 2);
  (void)ws_size; (void)in_sizes; (void)n_in; (void)out_size;

  dim3 blk(256);
  const float sscale = 0.08838834764831845f;   // 1/sqrt(128)

  // -- weight transposes (row-major f32 -> column-major bf16) --
  k_tr_f32_bf16<<<dim3((CC * 3 * CC + 255) / 256, 1), blk, 0, stream>>>(w_qkv,  wqkvT,  CC,   3 * CC, 0, 0);
  k_tr_f32_bf16<<<dim3((CC * CC     + 255) / 256, 1), blk, 0, stream>>>(w_proj, wprojT, CC,   CC,     0, 0);
  k_tr_f32_bf16<<<dim3((CC * HIDD   + 255) / 256, 1), blk, 0, stream>>>(w_fc1,  wfc1T,  CC,   HIDD,   0, 0);
  k_tr_f32_bf16<<<dim3((CC * HIDD   + 255) / 256, 1), blk, 0, stream>>>(w_gcn,  wgcnT,  CC,   HIDD,   0, 0);
  k_tr_f32_bf16<<<dim3((HIDD * CC   + 255) / 256, 1), blk, 0, stream>>>(w_fc2,  wfc2T,  HIDD, CC,     0, 0);

  // -- masks / adjacency to bf16 (row + transposed) --
  k_build_a <<<dim3((BLL + 255) / 256), blk, 0, stream>>>(rel, ab, aTb);
  k_cvt_pair<<<dim3((BLL + 255) / 256), blk, 0, stream>>>(adj, adjb, adjTb);

  // -- LN1 -> bf16, qkv GEMM, split heads --
  k_ln<<<dim3(ROWS), blk, 0, stream>>>(x, ln1g, ln1b, xn);
  k_gemm<false, false><<<dim3(3 * CC / 128, ROWS / 128, 1), blk, 0, stream>>>(
      xn, wqkvT, nullptr, qkvbuf, (int)ROWS, 3 * CC, CC, 1.f, 0, 0, 0, 0, 1, 3 * CC);
  k_split_qkv<<<dim3((BLC + 255) / 256), blk, 0, stream>>>(qkvbuf, qb, kb, vTb);

  // -- scores: q @ k^T (k row-major IS col-major of k^T) --
  k_gemm<false, false><<<dim3(LL / 128, LL / 128, BB * HH), blk, 0, stream>>>(
      qb, kb, nullptr, score, LL, LL, HSZ, sscale,
      (long)LL * HSZ, (long)LL * HSZ, (long)LL * LL, 0, 1, LL);

  // -- pe products: aT@a and a@aT (both operands come from a/aT buffers) --
  k_gemm<false, false><<<dim3(LL / 128, LL / 128, BB), blk, 0, stream>>>(
      aTb, aTb, nullptr, prod1, LL, LL, LL, 1.f,
      (long)LL * LL, (long)LL * LL, (long)LL * LL, 0, 1, LL);
  k_gemm<false, false><<<dim3(LL / 128, LL / 128, BB), blk, 0, stream>>>(
      ab, ab, nullptr, prod2, LL, LL, LL, 1.f,
      (long)LL * LL, (long)LL * LL, (long)LL * LL, 0, 1, LL);
  k_build_pe<<<dim3((BLL + 255) / 256), blk, 0, stream>>>(rel, prod1, prod2, pe);

  // -- masked softmax -> bf16 attn --
  k_softmax<<<dim3(BB * HH * LL), blk, 0, stream>>>(score, pe, attnb);

  // -- attn @ v, written head-interleaved into [B,L,C] bf16 --
  k_gemm<true, false><<<dim3(HSZ / 128, LL / 128, BB * HH), blk, 0, stream>>>(
      attnb, vTb, nullptr, aout, LL, HSZ, LL, 1.f,
      (long)LL * LL, (long)HSZ * LL, (long)LL * CC, (long)HSZ, HH, CC);

  // -- proj + residual: x1 = x + aout @ w_proj --
  k_gemm<false, true><<<dim3(CC / 128, ROWS / 128, 1), blk, 0, stream>>>(
      aout, wprojT, x, x1, (int)ROWS, CC, CC, 1.f, 0, 0, 0, 0, 1, CC);

  // -- LN2, fc1, gcn --
  k_ln<<<dim3(ROWS), blk, 0, stream>>>(x1, ln2g, ln2b, xn2);
  k_gemm<false, false><<<dim3(HIDD / 128, ROWS / 128, 1), blk, 0, stream>>>(
      xn2, wfc1T, nullptr, hbuf, (int)ROWS, HIDD, CC, 1.f, 0, 0, 0, 0, 1, HIDD);
  k_gemm<false, false><<<dim3(HIDD / 128, ROWS / 128, 1), blk, 0, stream>>>(
      xn2, wgcnT, nullptr, gbuf, (int)ROWS, HIDD, CC, 1.f, 0, 0, 0, 0, 1, HIDD);

  // -- transpose g per batch: [512,2048] f32 -> [2048,512] bf16 --
  k_tr_f32_bf16<<<dim3((LL * HIDD + 255) / 256, BB), blk, 0, stream>>>(
      gbuf, gT, LL, HIDD, (long)LL * HIDD, (long)HIDD * LL);

  // -- h[:, :1024] += adj @ g1 ; h[:, 1024:] += adj^T @ g2 (in-place f32 acc) --
  k_gemm<false, true><<<dim3((HIDD / 2) / 128, LL / 128, BB), blk, 0, stream>>>(
      adjb, gT, hbuf, hbuf, LL, HIDD / 2, LL, 1.f,
      (long)LL * LL, (long)HIDD * LL, (long)LL * HIDD, 0, 1, HIDD);
  k_gemm<false, true><<<dim3((HIDD / 2) / 128, LL / 128, BB), blk, 0, stream>>>(
      adjTb, gT + (size_t)(HIDD / 2) * LL, hbuf + HIDD / 2, hbuf + HIDD / 2, LL, HIDD / 2, LL, 1.f,
      (long)LL * LL, (long)HIDD * LL, (long)LL * HIDD, 0, 1, HIDD);

  // -- relu -> bf16, fc2 + residual straight into d_out --
  k_relu_cvt<<<dim3((ROWS * HIDD + 255) / 256), blk, 0, stream>>>(hbuf, hb, ROWS * HIDD);
  k_gemm<false, true><<<dim3(CC / 128, ROWS / 128, 1), blk, 0, stream>>>(
      hb, wfc2T, x1, out, (int)ROWS, CC, HIDD, 1.f, 0, 0, 0, 0, 1, CC);
}
